// DiffusionModule_11055245820602
// MI455X (gfx1250) — matrix operand order repaired and checked by reading, not compiled, not run on hardware
//
#include <hip/hip_runtime.h>
#include <hip/hip_bf16.h>

typedef _Float16 half_t;
typedef __attribute__((ext_vector_type(16))) _Float16 v16h;
typedef __attribute__((ext_vector_type(8)))  float    v8f;
typedef uint32_t u32x4 __attribute__((ext_vector_type(4)));
typedef uint32_t u32x8 __attribute__((ext_vector_type(8)));

static constexpr int NTOK = 2048;   // N
static constexpr int CDIM = 128;    // C == CS
static constexpr int C2   = 256;    // 2C
static constexpr int CP   = 16;
static constexpr int NLAY = 3;
static constexpr float EPSF = 1e-5f;

// ---------------- WMMA fragment loaders (contiguous k order, k = 16*(lane/16) + s) -------------

// A fragment 16x32 f16, A row-major [m][k], lda in elements.
__device__ inline v16h load_a_h(const half_t* A, int lda, int lane) {
  const half_t* p = A + (size_t)(lane & 15) * lda + ((lane >> 4) << 4);
  v16h r;
#pragma unroll
  for (int i = 0; i < 8; ++i) r[i] = p[i];
#pragma unroll
  for (int i = 0; i < 8; ++i) r[8 + i] = p[8 + i];
  return r;
}

__device__ inline v16h load_a_f32(const float* A, int lda, int lane) {
  const float* p = A + (size_t)(lane & 15) * lda + ((lane >> 4) << 4);
  v16h r;
#pragma unroll
  for (int i = 0; i < 8; ++i) r[i] = (half_t)p[i];
#pragma unroll
  for (int i = 0; i < 8; ++i) r[8 + i] = (half_t)p[8 + i];
  return r;
}

// B fragment 32x16 f16 from transposed storage Bt[n][k] (k contiguous).
__device__ inline v16h load_b_h(const half_t* Bt, int ldb, int lane) {
  const half_t* p = Bt + (size_t)(lane & 15) * ldb + ((lane >> 4) << 4);
  v16h r;
#pragma unroll
  for (int i = 0; i < 16; ++i) r[i] = p[i];
  return r;
}

// B fragment 32x16 f16 from k-major storage B[k][n] (n contiguous, stride ld per k)
__device__ inline v16h load_b_k(const half_t* B, int ld, int lane) {
  const half_t* p = B + (size_t)((lane >> 4) << 4) * ld + (lane & 15);
  v16h r;
#pragma unroll
  for (int i = 0; i < 16; ++i) r[i] = p[(size_t)i * ld];
  return r;
}

// ---------------- Tensor Data Mover: 2-D tile (dword rows) global -> LDS -------
// Descriptor per CDNA5 ISA ch.8: group0 {count,lds,global,type}, group1 {dims}.
// row_dwords: row length in 4-byte units; nrows: tile rows; tensor_rows: rows
// available from the tile start (OOB bound). Issued per-wave; EXEC ignored.
__device__ inline void tdm_load_2d(uint32_t lds_off, const void* gaddr,
                                   uint32_t row_dwords, uint32_t nrows,
                                   uint32_t tensor_rows) {
  uint64_t ga = (uint64_t)(uintptr_t)gaddr;
  u32x4 g0;
  g0[0] = 1u;                                      // count=1 (valid user D#)
  g0[1] = lds_off;                                 // lds_addr (bytes)
  g0[2] = (uint32_t)ga;                            // global_addr[31:0]
  g0[3] = (uint32_t)((ga >> 32) & 0x1FFFFFFu)      // global_addr[56:32]
          | (2u << 30);                            // type=2 ("image")
  u32x8 g1;
  g1[0] = (2u << 16);                              // workgroup_mask=0, data_size=4B
  g1[1] = (row_dwords & 0xFFFFu) << 16;            // tensor_dim0[15:0]
  g1[2] = (row_dwords >> 16) | ((tensor_rows & 0xFFFFu) << 16);  // dim0 hi | dim1 lo
  g1[3] = (tensor_rows >> 16) | ((row_dwords & 0xFFFFu) << 16);  // dim1 hi | tile_dim0
  g1[4] = nrows & 0xFFFFu;                         // tile_dim1 (tile_dim2=0)
  g1[5] = row_dwords;                              // tensor_dim0_stride[31:0]
  g1[6] = 0u;                                      // stride hi | dim1_stride lo
  g1[7] = 0u;
  asm volatile("tensor_load_to_lds %0, %1" :: "s"(g0), "s"(g1) : "memory");
}

// ---------------- generic WMMA GEMM: out[m][n] = act((A@Bt^T + bias)*scale) * aux ----
template <int NK>
__global__ __launch_bounds__(256) void gemm_kernel(
    const half_t* __restrict__ A, const half_t* __restrict__ Bt,
    const float* __restrict__ bias, const float* __restrict__ aux,
    float scale, int act,
    float* __restrict__ outF, half_t* __restrict__ outH, int Nc)
{
  const int K = NK * 32;
  const int bm = blockIdx.x * 128, bn = blockIdx.y * 128;
  const int wave = threadIdx.x >> 5, lane = threadIdx.x & 31;
  const int m0 = bm + wave * 16;
  const int mo = (lane >> 4) * 8, ncol = lane & 15;

  v16h afr[NK];
#pragma unroll
  for (int kk = 0; kk < NK; ++kk)
    afr[kk] = load_a_h(A + (size_t)m0 * K + kk * 32, K, lane);

  for (int nt = 0; nt < 8; ++nt) {
    const int n0 = bn + nt * 16;
    if (nt + 1 < 8)  // hint next weight tile into cache (global_prefetch_b8)
      __builtin_prefetch(Bt + (size_t)(n0 + 16) * K, 0, 3);
    v8f acc = {};
#pragma unroll
    for (int kk = 0; kk < NK; ++kk) {
      v16h b = load_b_h(Bt + (size_t)n0 * K + kk * 32, K, lane);
      acc = __builtin_amdgcn_wmma_f32_16x16x32_f16(false, afr[kk], false, b,
                                                   (short)0, acc, false, false);
    }
    const float bv = bias ? bias[n0 + ncol] : 0.0f;
#pragma unroll
    for (int r = 0; r < 8; ++r) {
      const int m = m0 + mo + r, n = n0 + ncol;
      float v = (acc[r] + bv) * scale;
      if (act == 1) v = 1.0f / (1.0f + __expf(-v));
      const size_t idx = (size_t)m * Nc + n;
      if (aux)  v *= aux[idx];
      if (outF) outF[idx] = v;
      if (outH) outH[idx] = (half_t)v;
    }
  }
}

// ---------------- small helpers -------------------------------------------------

__device__ inline float block_sum_128(float v, float* buf) {
#pragma unroll
  for (int off = 16; off > 0; off >>= 1) v += __shfl_xor(v, off, 32);
  __syncthreads();
  if ((threadIdx.x & 31) == 0) buf[threadIdx.x >> 5] = v;
  __syncthreads();
  return buf[0] + buf[1] + buf[2] + buf[3];
}

// transpose fp32 W[k][n] -> fp16 Wt[n][k]
__global__ void transpose_to_h_kernel(const float* __restrict__ W,
                                      half_t* __restrict__ Wt, int K, int N) {
  int i = blockIdx.x * 256 + threadIdx.x;
  if (i >= K * N) return;
  int k = i / N, n = i - k * N;
  Wt[(size_t)n * K + k] = (half_t)W[i];
}

// sn = LN(s) * ln_w  (per layer, fp16) ; s16 = fp16(s)
__global__ __launch_bounds__(128) void prep_s_kernel(
    const float* __restrict__ s, const float* __restrict__ ln1w,
    const float* __restrict__ ln2w, half_t* __restrict__ s16,
    half_t* __restrict__ sn1, half_t* __restrict__ sn2)
{
  __shared__ float buf[4];
  const int r = blockIdx.x, c = threadIdx.x;
  const size_t idx = (size_t)r * CDIM + c;
  float x = s[idx];
  float mu = block_sum_128(x, buf) * (1.0f / 128.0f);
  float d = x - mu;
  float var = block_sum_128(d * d, buf) * (1.0f / 128.0f);
  float ln = d * rsqrtf(var + EPSF);
  s16[idx] = (half_t)x;
  for (int l = 0; l < NLAY; ++l) {
    sn1[(size_t)l * NTOK * CDIM + idx] = (half_t)(ln * ln1w[l * CDIM + c]);
    sn2[(size_t)l * NTOK * CDIM + idx] = (half_t)(ln * ln2w[l * CDIM + c]);
  }
}

// ah = g1 * LN(a) + g2
__global__ __launch_bounds__(128) void ew_adaln_kernel(
    const float* __restrict__ a, const float* __restrict__ g1,
    const float* __restrict__ g2, float* __restrict__ outF,
    half_t* __restrict__ outH)
{
  __shared__ float buf[4];
  const int r = blockIdx.x, c = threadIdx.x;
  const size_t idx = (size_t)r * CDIM + c;
  float x = a[idx];
  float mu = block_sum_128(x, buf) * (1.0f / 128.0f);
  float d = x - mu;
  float var = block_sum_128(d * d, buf) * (1.0f / 128.0f);
  float ln = d * rsqrtf(var + EPSF);
  float o = g1[idx] * ln + g2[idx];
  if (outF) outF[idx] = o;
  outH[idx] = (half_t)o;
}

// bb = silu(b1) * b2   (N x 2C)
__global__ void ew_bb_kernel(const float* __restrict__ b1,
                             const float* __restrict__ b2,
                             half_t* __restrict__ bb, int total) {
  int i = blockIdx.x * 256 + threadIdx.x;
  if (i >= total) return;
  float x = b1[i];
  bb[i] = (half_t)((x / (1.0f + __expf(-x))) * b2[i]);
}

// a_new = attn + tgate * th * t3
__global__ void ew_final_kernel(const float* __restrict__ attn,
                                const float* __restrict__ tg,
                                const float* __restrict__ th,
                                const float* __restrict__ t3,
                                float* __restrict__ aout, int total) {
  int i = blockIdx.x * 256 + threadIdx.x;
  if (i >= total) return;
  aout[i] = attn[i] + tg[i] * th[i] * t3[i];
}

// ---------------- local banded attention (one 32-query block per workgroup) -----
// Covered rows l = 32k+1..32k+31, key window m in [32k-47, 32k+79], j = m-(32k-48)
// in [1,127]. Masked entries get -1e30 (exp underflows to exact 0, matching the
// reference's fp32 behaviour for beta=-1e10). Q/K/V tiles staged via TDM for
// interior blocks, overlapped with the zb (pair-bias) computation.
__global__ __launch_bounds__(256) void attn_local_kernel(
    const half_t* __restrict__ q16, const half_t* __restrict__ k16,
    const half_t* __restrict__ v16, const float* __restrict__ z,
    const float* __restrict__ lnz_w, const float* __restrict__ lnz_b,
    const float* __restrict__ zinj_w, const float* __restrict__ gct,
    half_t* __restrict__ o16)
{
  __shared__ half_t Qs[32 * 128];
  __shared__ half_t Ks[128 * 128];
  __shared__ half_t Vs[128 * 128];
  __shared__ float  Sb[4 * 32 * 128];   // zb -> scores -> probs
  __shared__ float  WZ[16 * 4];
  __shared__ float  BZ[4];

  const int tid = threadIdx.x;
  const int wave = tid >> 5, lane = tid & 31;
  const int qbase = blockIdx.x * 32;
  const int kbase = qbase - 48;
  const bool interior = (kbase >= 0) && (kbase + 128 <= NTOK);

  if (tid < 64) WZ[tid] = lnz_w[tid >> 2] * zinj_w[tid];          // WZ[c*4+h]
  if (tid < 4) {
    float b = 0.0f;
    for (int c = 0; c < 16; ++c) b += lnz_b[c] * zinj_w[c * 4 + tid];
    BZ[tid] = b;
  }

  if (interior) {
    // async tile DMA: rows land exactly as Ks[j][*] (m = kbase + j).
    if (wave == 0) {
      tdm_load_2d((uint32_t)(uintptr_t)Qs, q16 + (size_t)qbase * 128, 64, 32,
                  (uint32_t)(NTOK - qbase));
      tdm_load_2d((uint32_t)(uintptr_t)Ks, k16 + (size_t)kbase * 128, 64, 128,
                  (uint32_t)(NTOK - kbase));
      tdm_load_2d((uint32_t)(uintptr_t)Vs, v16 + (size_t)kbase * 128, 64, 128,
                  (uint32_t)(NTOK - kbase));
    }
  } else {
    for (int i = tid; i < 32 * 128; i += 256)
      Qs[i] = q16[(size_t)(qbase + (i >> 7)) * 128 + (i & 127)];
    for (int i = tid; i < 128 * 128; i += 256) {
      int j = i >> 7, cc = i & 127, m = kbase + j;
      bool valid = (j > 0) && (m >= 0) && (m < NTOK);
      Ks[i] = valid ? k16[(size_t)m * 128 + cc] : (half_t)0.0f;
      Vs[i] = valid ? v16[(size_t)m * 128 + cc] : (half_t)0.0f;
    }
  }
  __syncthreads();   // WZ/BZ ready (TDM still in flight for interior blocks)

  // pair bias zb[h][l][j] from LN(z) (folded per-head weights), masked -1e30.
  // Independent of Ks/Vs/Qs -> overlaps the TDM transfers.
  for (int t = tid; t < 32 * 128; t += 256) {
    int l = t >> 7, j = t & 127, m = kbase + j;
    bool valid = (j > 0) && (m >= 0) && (m < NTOK);
    float sh[4];
    if (valid) {
      const float* zp = z + ((size_t)(qbase + l) * NTOK + m) * CP;
      float xv[16], mu = 0.0f;
#pragma unroll
      for (int c = 0; c < 16; ++c) { xv[c] = zp[c]; mu += xv[c]; }
      mu *= (1.0f / 16.0f);
      float var = 0.0f;
#pragma unroll
      for (int c = 0; c < 16; ++c) { float d = xv[c] - mu; var += d * d; }
      float rs = rsqrtf(var * (1.0f / 16.0f) + EPSF);
      sh[0] = BZ[0]; sh[1] = BZ[1]; sh[2] = BZ[2]; sh[3] = BZ[3];
#pragma unroll
      for (int c = 0; c < 16; ++c) {
        float zn = (xv[c] - mu) * rs;
        sh[0] += zn * WZ[c * 4 + 0]; sh[1] += zn * WZ[c * 4 + 1];
        sh[2] += zn * WZ[c * 4 + 2]; sh[3] += zn * WZ[c * 4 + 3];
      }
    } else {
      sh[0] = sh[1] = sh[2] = sh[3] = -1e30f;
    }
#pragma unroll
    for (int h = 0; h < 4; ++h) Sb[(h * 32 + l) * 128 + j] = sh[h];
  }

  if (interior && wave == 0) __builtin_amdgcn_s_wait_tensorcnt(0);
  __syncthreads();   // Q/K/V tiles + zb all visible

  const int mo = (lane >> 4) * 8, ncol = lane & 15;

  // S = Q K^T + zb : 64 tiles (4 heads x 2 mtiles x 8 ktiles), zb as WMMA C input
  for (int i = 0; i < 8; ++i) {
    int t = wave * 8 + i;
    int h = t >> 4, mt = (t >> 3) & 1, kt = t & 7;
    v16h a = load_a_h(Qs + (mt * 16) * 128 + h * 32, 128, lane);
    v16h b = load_b_h(Ks + (kt * 16) * 128 + h * 32, 128, lane);
    v8f cfr;
#pragma unroll
    for (int r = 0; r < 8; ++r)
      cfr[r] = Sb[(h * 32 + mt * 16 + mo + r) * 128 + kt * 16 + ncol];
    cfr = __builtin_amdgcn_wmma_f32_16x16x32_f16(false, a, false, b, (short)0,
                                                 cfr, false, false);
#pragma unroll
    for (int r = 0; r < 8; ++r)
      Sb[(h * 32 + mt * 16 + mo + r) * 128 + kt * 16 + ncol] = cfr[r];
  }
  __syncthreads();

  // row softmax over 128 keys (128 rows = 4 heads x 32 queries)
  if (tid < 128) {
    float* row = Sb + (size_t)tid * 128;
    float mx = -1e38f;
    for (int j = 0; j < 128; ++j) mx = fmaxf(mx, row[j]);
    float sum = 0.0f;
    for (int j = 0; j < 128; ++j) { float e = __expf(row[j] - mx); row[j] = e; sum += e; }
    float rinv = 1.0f / sum;
    for (int j = 0; j < 128; ++j) row[j] *= rinv;
  }
  __syncthreads();

  // O = P V, gate with sigmoid(ah@g_w), store fp16
  for (int i = 0; i < 2; ++i) {
    int t = wave * 2 + i;
    int h = t >> 2, mt = (t >> 1) & 1, nt = t & 1;
    v8f acc = {};
#pragma unroll
    for (int kk = 0; kk < 4; ++kk) {
      v16h a = load_a_f32(Sb + (size_t)(h * 32 + mt * 16) * 128 + kk * 32, 128, lane);
      v16h b = load_b_k(Vs + (size_t)(kk * 32) * 128 + h * 32 + nt * 16, 128, lane);
      acc = __builtin_amdgcn_wmma_f32_16x16x32_f16(false, a, false, b, (short)0,
                                                   acc, false, false);
    }
#pragma unroll
    for (int r = 0; r < 8; ++r) {
      int row = qbase + mt * 16 + mo + r;
      int col = h * 32 + nt * 16 + ncol;
      size_t idx = (size_t)row * 128 + col;
      o16[idx] = (half_t)(acc[r] * gct[idx]);
    }
  }
}

// rows l % 32 == 0: scores all exactly -1e10 in fp32 -> uniform softmax -> o = mean(v)
__global__ __launch_bounds__(128) void vmean_kernel(const half_t* __restrict__ v16,
                                                    float* __restrict__ vmean) {
  int c = threadIdx.x;
  float s = 0.0f;
  for (int r = 0; r < NTOK; ++r) s += (float)v16[(size_t)r * 128 + c];
  vmean[c] = s * (1.0f / (float)NTOK);
}

__global__ __launch_bounds__(128) void global_rows_kernel(
    const float* __restrict__ vmean, const float* __restrict__ gct,
    half_t* __restrict__ o16) {
  int row = blockIdx.x * 32, c = threadIdx.x;
  size_t idx = (size_t)row * 128 + c;
  o16[idx] = (half_t)(vmean[c] * gct[idx]);
}

// ---------------- host orchestration -------------------------------------------

extern "C" void kernel_launch(void* const* d_in, const int* in_sizes, int n_in,
                              void* d_out, int out_size, void* d_ws, size_t ws_size,
                              hipStream_t stream)
{
  const float* a_in       = (const float*)d_in[0];
  const float* s_in       = (const float*)d_in[1];
  const float* z_in       = (const float*)d_in[2];
  const float* ada1_ln_w  = (const float*)d_in[4];
  const float* ada1_lin_w = (const float*)d_in[5];
  const float* ada1_lin_b = (const float*)d_in[6];
  const float* ada1_nb_w  = (const float*)d_in[7];
  const float* lnz_w      = (const float*)d_in[8];
  const float* lnz_b      = (const float*)d_in[9];
  const float* zinj_w     = (const float*)d_in[10];
  const float* q_w        = (const float*)d_in[11];
  const float* q_b        = (const float*)d_in[12];
  const float* k_w        = (const float*)d_in[13];
  const float* v_w        = (const float*)d_in[14];
  const float* g_w        = (const float*)d_in[15];
  const float* o_w        = (const float*)d_in[16];
  const float* agate_w    = (const float*)d_in[17];
  const float* agate_b    = (const float*)d_in[18];
  const float* ada2_ln_w  = (const float*)d_in[19];
  const float* ada2_lin_w = (const float*)d_in[20];
  const float* ada2_lin_b = (const float*)d_in[21];
  const float* ada2_nb_w  = (const float*)d_in[22];
  const float* lin1_w     = (const float*)d_in[23];
  const float* lin2_w     = (const float*)d_in[24];
  const float* lin3_w     = (const float*)d_in[25];
  const float* tgate_w    = (const float*)d_in[26];
  const float* tgate_b    = (const float*)d_in[27];

  char* wsb = (char*)d_ws;
  size_t off = 0;
  auto alloc = [&](size_t bytes) -> void* {
    void* p = wsb + off;
    off += (bytes + 255) & ~(size_t)255;
    return p;
  };

  static constexpr size_t OFF_A1LIN = 0, OFF_A1NB = 16384, OFF_QW = 32768,
      OFF_KW = 49152, OFF_VW = 65536, OFF_GW = 81920, OFF_OW = 98304,
      OFF_AG = 114688, OFF_A2LIN = 131072, OFF_A2NB = 147456, OFF_TG = 163840,
      OFF_L1 = 180224, OFF_L2 = 212992, OFF_L3 = 245760, LAYER_W = 278528;

  half_t* w16   = (half_t*)alloc(NLAY * LAYER_W * sizeof(half_t));
  half_t* s16   = (half_t*)alloc((size_t)NTOK * CDIM * 2);
  half_t* sn1   = (half_t*)alloc((size_t)NLAY * NTOK * CDIM * 2);
  half_t* sn2   = (half_t*)alloc((size_t)NLAY * NTOK * CDIM * 2);
  float*  agp   = (float*)alloc((size_t)NLAY * NTOK * CDIM * 4);
  float*  tgp   = (float*)alloc((size_t)NLAY * NTOK * CDIM * 4);
  float*  acur  = (float*)alloc((size_t)NTOK * CDIM * 4);
  float*  g1f   = (float*)alloc((size_t)NTOK * CDIM * 4);
  float*  g2f   = (float*)alloc((size_t)NTOK * CDIM * 4);
  half_t* ah16  = (half_t*)alloc((size_t)NTOK * CDIM * 2);
  float*  thF   = (float*)alloc((size_t)NTOK * CDIM * 4);
  half_t* th16  = (half_t*)alloc((size_t)NTOK * CDIM * 2);
  half_t* q16b  = (half_t*)alloc((size_t)NTOK * CDIM * 2);
  half_t* k16b  = (half_t*)alloc((size_t)NTOK * CDIM * 2);
  half_t* v16b  = (half_t*)alloc((size_t)NTOK * CDIM * 2);
  float*  gctF  = (float*)alloc((size_t)NTOK * CDIM * 4);
  half_t* o16b  = (half_t*)alloc((size_t)NTOK * CDIM * 2);
  float*  attnF = (float*)alloc((size_t)NTOK * CDIM * 4);
  float*  b1f   = (float*)alloc((size_t)NTOK * C2 * 4);
  float*  b2f   = (float*)alloc((size_t)NTOK * C2 * 4);
  half_t* bb16  = (half_t*)alloc((size_t)NTOK * C2 * 2);
  float*  t3F   = (float*)alloc((size_t)NTOK * CDIM * 4);
  float*  vmean = (float*)alloc(CDIM * 4);

  auto tr = [&](const float* W, half_t* Wt, int K, int N) {
    int tot = K * N;
    transpose_to_h_kernel<<<(tot + 255) / 256, 256, 0, stream>>>(W, Wt, K, N);
  };
  auto gemm = [&](const half_t* A, const half_t* Bt, const float* bias,
                  const float* aux, float scl, int act, float* oF, half_t* oH,
                  int Nc, int K) {
    dim3 g(NTOK / 128, Nc / 128);
    if (K == 128)
      gemm_kernel<4><<<g, 256, 0, stream>>>(A, Bt, bias, aux, scl, act, oF, oH, Nc);
    else
      gemm_kernel<8><<<g, 256, 0, stream>>>(A, Bt, bias, aux, scl, act, oF, oH, Nc);
  };

  // --- prep: transpose/convert all weights to fp16 ---
  for (int l = 0; l < NLAY; ++l) {
    half_t* wl = w16 + (size_t)l * LAYER_W;
    tr(ada1_lin_w + (size_t)l * 128 * 128, wl + OFF_A1LIN, 128, 128);
    tr(ada1_nb_w  + (size_t)l * 128 * 128, wl + OFF_A1NB,  128, 128);
    tr(q_w        + (size_t)l * 128 * 128, wl + OFF_QW,    128, 128);
    tr(k_w        + (size_t)l * 128 * 128, wl + OFF_KW,    128, 128);
    tr(v_w        + (size_t)l * 128 * 128, wl + OFF_VW,    128, 128);
    tr(g_w        + (size_t)l * 128 * 128, wl + OFF_GW,    128, 128);
    tr(o_w        + (size_t)l * 128 * 128, wl + OFF_OW,    128, 128);
    tr(agate_w    + (size_t)l * 128 * 128, wl + OFF_AG,    128, 128);
    tr(ada2_lin_w + (size_t)l * 128 * 128, wl + OFF_A2LIN, 128, 128);
    tr(ada2_nb_w  + (size_t)l * 128 * 128, wl + OFF_A2NB,  128, 128);
    tr(tgate_w    + (size_t)l * 128 * 128, wl + OFF_TG,    128, 128);
    tr(lin1_w     + (size_t)l * 128 * 256, wl + OFF_L1,    128, 256);
    tr(lin2_w     + (size_t)l * 128 * 256, wl + OFF_L2,    128, 256);
    tr(lin3_w     + (size_t)l * 256 * 128, wl + OFF_L3,    256, 128);
  }

  // --- prep: s-dependent tensors (shared across the layer loop) ---
  prep_s_kernel<<<NTOK, 128, 0, stream>>>(s_in, ada1_ln_w, ada2_ln_w, s16, sn1, sn2);
  for (int l = 0; l < NLAY; ++l) {
    half_t* wl = w16 + (size_t)l * LAYER_W;
    gemm(s16, wl + OFF_AG, agate_b + l * CDIM, nullptr, 1.0f, 1,
         agp + (size_t)l * NTOK * CDIM, nullptr, CDIM, 128);
    gemm(s16, wl + OFF_TG, tgate_b + l * CDIM, nullptr, 1.0f, 1,
         tgp + (size_t)l * NTOK * CDIM, nullptr, CDIM, 128);
  }

  hipMemcpyAsync(acur, a_in, (size_t)NTOK * CDIM * 4, hipMemcpyDeviceToDevice, stream);

  const float qscale = 0.17677669529663687f;  // 1/sqrt(32)
  const int totC  = NTOK * CDIM;
  const int totC2 = NTOK * C2;

  for (int l = 0; l < NLAY; ++l) {
    half_t* wl = w16 + (size_t)l * LAYER_W;
    const half_t* sn1l = sn1 + (size_t)l * NTOK * CDIM;
    const half_t* sn2l = sn2 + (size_t)l * NTOK * CDIM;
    const float*  agpl = agp + (size_t)l * NTOK * CDIM;
    const float*  tgpl = tgp + (size_t)l * NTOK * CDIM;

    // --- AttentionPairBias ---
    gemm(sn1l, wl + OFF_A1LIN, ada1_lin_b + l * CDIM, nullptr, 1.0f, 1, g1f, nullptr, CDIM, 128);
    gemm(sn1l, wl + OFF_A1NB,  nullptr, nullptr, 1.0f, 0, g2f, nullptr, CDIM, 128);
    ew_adaln_kernel<<<NTOK, 128, 0, stream>>>(acur, g1f, g2f, nullptr, ah16);

    gemm(ah16, wl + OFF_QW, q_b + l * CDIM, nullptr, qscale, 0, nullptr, q16b, CDIM, 128);
    gemm(ah16, wl + OFF_KW, nullptr, nullptr, 1.0f, 0, nullptr, k16b, CDIM, 128);
    gemm(ah16, wl + OFF_VW, nullptr, nullptr, 1.0f, 0, nullptr, v16b, CDIM, 128);
    gemm(ah16, wl + OFF_GW, nullptr, nullptr, 1.0f, 1, gctF, nullptr, CDIM, 128);

    attn_local_kernel<<<NTOK / 32, 256, 0, stream>>>(
        q16b, k16b, v16b, z_in, lnz_w + l * CP, lnz_b + l * CP,
        zinj_w + l * CP * 4, gctF, o16b);
    vmean_kernel<<<1, 128, 0, stream>>>(v16b, vmean);
    global_rows_kernel<<<NTOK / 32, 128, 0, stream>>>(vmean, gctF, o16b);

    gemm(o16b, wl + OFF_OW, nullptr, agpl, 1.0f, 0, attnF, nullptr, CDIM, 128);

    // --- ConditionedTransitionBlock ---
    gemm(sn2l, wl + OFF_A2LIN, ada2_lin_b + l * CDIM, nullptr, 1.0f, 1, g1f, nullptr, CDIM, 128);
    gemm(sn2l, wl + OFF_A2NB,  nullptr, nullptr, 1.0f, 0, g2f, nullptr, CDIM, 128);
    ew_adaln_kernel<<<NTOK, 128, 0, stream>>>(acur, g1f, g2f, thF, th16);

    gemm(th16, wl + OFF_L1, nullptr, nullptr, 1.0f, 0, b1f, nullptr, C2, 128);
    gemm(th16, wl + OFF_L2, nullptr, nullptr, 1.0f, 0, b2f, nullptr, C2, 128);
    ew_bb_kernel<<<(totC2 + 255) / 256, 256, 0, stream>>>(b1f, b2f, bb16, totC2);
    gemm(bb16, wl + OFF_L3, nullptr, nullptr, 1.0f, 0, t3F, nullptr, CDIM, 256);

    ew_final_kernel<<<(totC + 255) / 256, 256, 0, stream>>>(attnF, tgpl, thF, t3F, acur, totC);
  }

  hipMemcpyAsync(d_out, acur, (size_t)NTOK * CDIM * 4, hipMemcpyDeviceToDevice, stream);
}
